// INN_Neuron_44976897523899
// MI455X (gfx1250) — compile-verified, matrix operand order checked by reading, not run on hardware
//
#include <hip/hip_runtime.h>
#include <hip/hip_bf16.h>
#include <cstdint>
#include <cstddef>

// ---------------------------------------------------------------------------
// INN_Neuron attention-LSTM cell, MI455X (gfx1250) implementation.
//
// Algebraic fusion: softmax is over n only and k/v projections are linear, so
//   scores[b,h,n] = states[b,n,:] . wkq[b,h,:],  wkq = (Wk_h^T q_h)/sqrt(DH)
//   ctx[b,h,:]    = sv[b,h,:] @ Wv_h^T + bv_h,   sv  = sum_n attn * states[b,n,:]
// -> all_states is streamed ONCE (1.07 GB ~ 46us @ 23.3 TB/s) with an online
// softmax; all GEMMs shrink to ~19 GFLOP and run on v_wmma_f32_16x16x32_bf16.
// ---------------------------------------------------------------------------

#define Dm   1024
#define DHm  512

typedef __attribute__((ext_vector_type(16))) __bf16 v16bf;
typedef __attribute__((ext_vector_type(8)))  float  v8f;

union V16B { v16bf v; uint4 q[2]; };
union V8F  { v8f   v; float f[8]; };

// ---- CDNA5 async global->LDS (ASYNCcnt path), with safe fallback ----------
#if defined(__AMDGCN__) && __has_builtin(__builtin_amdgcn_global_load_async_to_lds_b128)
#define ATHENA_ASYNC_LDS 1
typedef __attribute__((__vector_size__(16))) int v4i_;
typedef __attribute__((address_space(1))) v4i_ as1_v4i;
typedef __attribute__((address_space(3))) v4i_ as3_v4i;
#endif

__device__ __forceinline__ void g2l_b128(void* lds_dst, const void* gsrc) {
#ifdef ATHENA_ASYNC_LDS
  __builtin_amdgcn_global_load_async_to_lds_b128((as1_v4i*)gsrc, (as3_v4i*)lds_dst, 0, 0);
#else
  *(float4*)lds_dst = *(const float4*)gsrc;
#endif
}

#ifdef ATHENA_ASYNC_LDS
#if __has_builtin(__builtin_amdgcn_s_wait_asynccnt)
#define WAIT_ASYNC(n) __builtin_amdgcn_s_wait_asynccnt(n)
#else
#define WAIT_ASYNC(n) asm volatile("s_wait_asynccnt %0" :: "i"(n) : "memory")
#endif
#else
#define WAIT_ASYNC(n) ((void)0)
#endif

__device__ __forceinline__ float warp_sum32(float v) {
#pragma unroll
  for (int o = 16; o > 0; o >>= 1) v += __shfl_down(v, o, 32);
  return v;
}

__device__ __forceinline__ float sigmoidf_(float x) { return 1.0f / (1.0f + __expf(-x)); }

// ---------------------------------------------------------------------------
// bf16 WMMA GEMM:  C[M,N] = alpha * A[M,K] @ W[N,K]^T  (+ bias[n]) (+= C)
// A row-major bf16 (lda), W row-major [n][k] bf16 (ldw, i.e. x@W^T form).
// One 16x16 C tile per wave via v_wmma_f32_16x16x32_bf16; 8 waves -> 16x128
// workgroup tile. Per-lane operand fetch = 2x b128 loads per operand per
// K-step of 32, matching the 16-bit A/B VGPR layout (lanes 0-15: K 0-7/16-23,
// lanes 16-31: K 8-15/24-31; row/col = lane & 15).
// ---------------------------------------------------------------------------
__global__ __launch_bounds__(256) void k_gemm_bf16_wmma(
    const __hip_bfloat16* __restrict__ A, int lda,
    const __hip_bfloat16* __restrict__ W, int ldw,
    float* __restrict__ Cf, __hip_bfloat16* __restrict__ Cbf, int ldc,
    const float* __restrict__ bias,
    int M, int N, int K, float alpha, int accumulate)
{
  const int wave = threadIdx.x >> 5;
  const int lane = threadIdx.x & 31;
  const int r    = lane & 15;      // row (A) / col-row (W) within tile
  const int hs   = lane >> 4;      // half-select: which K sub-block this lane holds

  const int n0 = blockIdx.x * 128 + wave * 16;
  const int m0 = blockIdx.y * 16;
  if (n0 >= N || m0 >= M) return;

  const __hip_bfloat16* Ap = A + (size_t)(m0 + r) * lda + hs * 8;
  const __hip_bfloat16* Wp = W + (size_t)(n0 + r) * ldw + hs * 8;

  V8F acc;
#pragma unroll
  for (int i = 0; i < 8; ++i) acc.f[i] = 0.0f;

  for (int k = 0; k < K; k += 32) {
    V16B a, b;
    a.q[0] = *(const uint4*)(Ap + k);        // K = k + hs*8 + {0..7}
    a.q[1] = *(const uint4*)(Ap + k + 16);   // K = k + 16 + hs*8 + {0..7}
    b.q[0] = *(const uint4*)(Wp + k);
    b.q[1] = *(const uint4*)(Wp + k + 16);
    __builtin_prefetch(Wp + k + 128, 0, 0);  // global_prefetch_b8
    acc.v = __builtin_amdgcn_wmma_f32_16x16x32_bf16(
        /*neg_a=*/false, a.v, /*neg_b=*/false, b.v,
        /*c_mod=*/(short)0, acc.v, /*reuse_a=*/false, /*reuse_b=*/false);
  }

  const int col = n0 + r;
#pragma unroll
  for (int i = 0; i < 8; ++i) {
    const int row = m0 + i + hs * 8;         // C/D layout: VGPR i -> M=i (+8 for hi lanes)
    float v = acc.f[i] * alpha;
    if (bias) v += bias[col];
    const size_t idx = (size_t)row * ldc + col;
    if (accumulate) v += Cf[idx];
    if (Cf)  Cf[idx]  = v;
    if (Cbf) Cbf[idx] = __float2bfloat16(v);
  }
}

// ---------------------------------------------------------------------------
// Elementwise conversion / packing kernels
// ---------------------------------------------------------------------------
__global__ void k_cvt_f32_bf16(const float* __restrict__ src,
                               __hip_bfloat16* __restrict__ dst, size_t n) {
  for (size_t i = (size_t)blockIdx.x * blockDim.x + threadIdx.x; i < n;
       i += (size_t)gridDim.x * blockDim.x)
    dst[i] = __float2bfloat16(src[i]);
}

// wkT[h][e][d] = Wk[h*DH + d, e] = in_proj_w[(D + h*DH + d)*D + e]
__global__ void k_transpose_wk(const float* __restrict__ in_proj_w,
                               __hip_bfloat16* __restrict__ wkT) {
  const size_t total = (size_t)2 * Dm * DHm;
  for (size_t i = (size_t)blockIdx.x * blockDim.x + threadIdx.x; i < total;
       i += (size_t)gridDim.x * blockDim.x) {
    const int d = (int)(i % DHm);
    const size_t t = i / DHm;
    const int e = (int)(t % Dm);
    const int hh = (int)(t / Dm);
    wkT[i] = __float2bfloat16(in_proj_w[(size_t)(Dm + hh * DHm + d) * Dm + e]);
  }
}

// rnn_in = concat(context, external_input) in bf16
__global__ void k_build_rnnin(const float* __restrict__ context,
                              const float* __restrict__ ext,
                              __hip_bfloat16* __restrict__ rnnin, int B) {
  const size_t total = (size_t)B * 2 * Dm;
  for (size_t i = (size_t)blockIdx.x * blockDim.x + threadIdx.x; i < total;
       i += (size_t)gridDim.x * blockDim.x) {
    const int j = (int)(i % (2 * Dm));
    const size_t b = i / (2 * Dm);
    const float v = (j < Dm) ? context[b * Dm + j] : ext[b * Dm + (j - Dm)];
    rnnin[i] = __float2bfloat16(v);
  }
}

// ---------------------------------------------------------------------------
// Fused streaming attention (the HBM-bound part).
// 1 workgroup per batch b, 256 threads = 8 wave32.
// Streams all_states[b] once through double-buffered LDS tiles (4 rows x D f32,
// filled with async global->LDS b128), online softmax over n with per-thread
// f32 accumulators of the weighted state sum sv[b,h,:].
// ---------------------------------------------------------------------------
__global__ __launch_bounds__(256) void k_attn_stream(
    const float* __restrict__ all_states,   // [B,N,D] f32
    const float* __restrict__ wkq,          // [B,2,D] f32 (score vectors, pre-scaled)
    __hip_bfloat16* __restrict__ sv,        // [B,2,D] bf16 out
    int Nn)
{
  constexpr int NT = 4;
  const int b = blockIdx.x;
  const int t = threadIdx.x;
  const int wave = t >> 5, lane = t & 31;

  __shared__ float tile[2][NT * Dm];        // 32 KB double buffer
  __shared__ float wq_s[2 * Dm];            // 8 KB per-batch score vectors
  __shared__ float s_sc[2][NT], s_w[2][NT], s_r[2], s_m[2], s_l[2];

  const float* S = all_states + (size_t)b * Nn * Dm;
  for (int i = t; i < 2 * Dm; i += 256) wq_s[i] = wkq[(size_t)b * 2 * Dm + i];
  if (t < 2) { s_m[t] = -3.0e38f; s_l[t] = 0.0f; }

  float acc[8];
#pragma unroll
  for (int i = 0; i < 8; ++i) acc[i] = 0.0f;
  const int hm = t >> 7;                    // head owned by this thread (0/1)
  const int e0 = (t & 127) * 8;             // 8 contiguous dims per thread

  const int ntiles = Nn / NT;

  // prime tile 0
#pragma unroll
  for (int i = 0; i < 4; ++i) {
    const int chunk = t + 256 * i;          // 16B chunks, 1024 per tile
    g2l_b128(&tile[0][chunk * 4], S + (size_t)chunk * 4);
  }

  for (int ti = 0; ti < ntiles; ++ti) {
    const int buf = ti & 1;
    if (ti + 1 < ntiles) {
      const float* src = S + (size_t)(ti + 1) * NT * Dm;
#pragma unroll
      for (int i = 0; i < 4; ++i) {
        const int chunk = t + 256 * i;
        g2l_b128(&tile[buf ^ 1][chunk * 4], src + (size_t)chunk * 4);
      }
      WAIT_ASYNC(4);                        // current tile done, next 4 in flight
    } else {
      WAIT_ASYNC(0);
    }
    __syncthreads();

    // scores: wave w handles (n = w>>1, h = w&1)
    {
      const int n  = wave >> 1;
      const int hh = wave & 1;
      const float* row = &tile[buf][n * Dm];
      const float* wv  = &wq_s[hh * Dm];
      float p = 0.0f;
      for (int e = lane; e < Dm; e += 32) p += row[e] * wv[e];
      p = warp_sum32(p);
      if (lane == 0) s_sc[hh][n] = p;
    }
    __syncthreads();

    // online softmax bookkeeping (one thread per head)
    if (t < 2) {
      const int hh = t;
      const float m_old = s_m[hh];
      float mx = m_old;
#pragma unroll
      for (int n = 0; n < NT; ++n) mx = fmaxf(mx, s_sc[hh][n]);
      const float rsc = __expf(m_old - mx);
      float sw = 0.0f;
#pragma unroll
      for (int n = 0; n < NT; ++n) {
        const float w = __expf(s_sc[hh][n] - mx);
        s_w[hh][n] = w;
        sw += w;
      }
      s_l[hh] = s_l[hh] * rsc + sw;
      s_m[hh] = mx;
      s_r[hh] = rsc;
    }
    __syncthreads();

    // accumulate weighted state sum
    {
      const float rsc = s_r[hm];
      const float w0 = s_w[hm][0], w1 = s_w[hm][1], w2 = s_w[hm][2], w3 = s_w[hm][3];
#pragma unroll
      for (int i = 0; i < 8; ++i) {
        const int e = e0 + i;
        const float v = w0 * tile[buf][0 * Dm + e] + w1 * tile[buf][1 * Dm + e] +
                        w2 * tile[buf][2 * Dm + e] + w3 * tile[buf][3 * Dm + e];
        acc[i] = acc[i] * rsc + v;
      }
    }
    __syncthreads();                        // protect buf before it is refilled
  }

  const float inv_l = 1.0f / s_l[hm];
#pragma unroll
  for (int i = 0; i < 8; ++i)
    sv[(size_t)b * 2 * Dm + (size_t)hm * Dm + e0 + i] =
        __float2bfloat16(acc[i] * inv_l);
}

// ---------------------------------------------------------------------------
// LSTM cell + LayerNorm epilogue. 1 WG per batch row, 256 threads.
// out[0 : B*D]       = layer_norm(new_h)
// out[B*D : 2*B*D]   = new_c
// ---------------------------------------------------------------------------
__global__ __launch_bounds__(256) void k_lstm_ln(
    const float* __restrict__ gates,   // [B, 4D] (i,f,g,o)
    const float* __restrict__ c,       // [B, D]
    const float* __restrict__ ln_w, const float* __restrict__ ln_b,
    float* __restrict__ out, int B)
{
  const int b = blockIdx.x;
  const int t = threadIdx.x;
  const size_t g0 = (size_t)b * 4 * Dm;

  float nh[4];
  float s1 = 0.0f, s2 = 0.0f;
#pragma unroll
  for (int i = 0; i < 4; ++i) {
    const int j = t + 256 * i;
    const float ig = gates[g0 + j];
    const float fg = gates[g0 + Dm + j];
    const float gg = gates[g0 + 2 * Dm + j];
    const float og = gates[g0 + 3 * Dm + j];
    const float cc = c[(size_t)b * Dm + j];
    const float nc = sigmoidf_(fg) * cc + sigmoidf_(ig) * tanhf(gg);
    out[(size_t)B * Dm + (size_t)b * Dm + j] = nc;
    const float hh = sigmoidf_(og) * tanhf(nc);
    nh[i] = hh;
    s1 += hh;
    s2 += hh * hh;
  }

  __shared__ float red1[8], red2[8];
  __shared__ float s_mu, s_rstd;
  s1 = warp_sum32(s1);
  s2 = warp_sum32(s2);
  const int wave = t >> 5, lane = t & 31;
  if (lane == 0) { red1[wave] = s1; red2[wave] = s2; }
  __syncthreads();
  if (t == 0) {
    float a = 0.0f, q = 0.0f;
#pragma unroll
    for (int i = 0; i < 8; ++i) { a += red1[i]; q += red2[i]; }
    const float mu = a / (float)Dm;
    const float var = q / (float)Dm - mu * mu;
    s_mu = mu;
    s_rstd = rsqrtf(var + 1e-5f);
  }
  __syncthreads();
#pragma unroll
  for (int i = 0; i < 4; ++i) {
    const int j = t + 256 * i;
    out[(size_t)b * Dm + j] = (nh[i] - s_mu) * s_rstd * ln_w[j] + ln_b[j];
  }
}

// ---------------------------------------------------------------------------
// Host orchestration
// ---------------------------------------------------------------------------
extern "C" void kernel_launch(void* const* d_in, const int* in_sizes, int n_in,
                              void* d_out, int out_size, void* d_ws, size_t ws_size,
                              hipStream_t stream) {
  const float* h_in       = (const float*)d_in[0];
  const float* c_in       = (const float*)d_in[1];
  const float* all_states = (const float*)d_in[2];
  const float* ext_in     = (const float*)d_in[3];
  const float* in_proj_w  = (const float*)d_in[4];
  const float* in_proj_b  = (const float*)d_in[5];
  const float* out_proj_w = (const float*)d_in[6];
  const float* out_proj_b = (const float*)d_in[7];
  const float* w_ih       = (const float*)d_in[8];
  const float* b_ih       = (const float*)d_in[9];
  const float* w_hh       = (const float*)d_in[10];
  const float* b_hh       = (const float*)d_in[11];
  const float* ln_w       = (const float*)d_in[12];
  const float* ln_b       = (const float*)d_in[13];
  float* out = (float*)d_out;

  const int B = in_sizes[0] / Dm;              // 512
  const int N = in_sizes[2] / in_sizes[0];     // 512

  using bf = __hip_bfloat16;
  char* ws = (char*)d_ws;
  size_t off = 0;
  auto alloc = [&](size_t bytes) -> void* {
    off = (off + 255) & ~(size_t)255;
    void* p = ws + off;
    off += bytes;
    return p;
  };

  bf* hbf      = (bf*)alloc((size_t)B * Dm * 2);
  bf* wqbf     = (bf*)alloc((size_t)Dm * Dm * 2);
  bf* wkT      = (bf*)alloc((size_t)2 * Dm * DHm * 2);
  bf* wvbf     = (bf*)alloc((size_t)Dm * Dm * 2);
  bf* wobf     = (bf*)alloc((size_t)Dm * Dm * 2);
  bf* wihbf    = (bf*)alloc((size_t)4 * Dm * 2 * Dm * 2);
  bf* whhbf    = (bf*)alloc((size_t)4 * Dm * Dm * 2);
  bf* qbf      = (bf*)alloc((size_t)B * Dm * 2);
  float* wkq   = (float*)alloc((size_t)B * 2 * Dm * 4);
  bf* svbf     = (bf*)alloc((size_t)B * 2 * Dm * 2);
  float* ctxf  = (float*)alloc((size_t)B * Dm * 4);
  bf* ctxbf    = (bf*)alloc((size_t)B * Dm * 2);
  float* conf  = (float*)alloc((size_t)B * Dm * 4);
  bf* rnninbf  = (bf*)alloc((size_t)B * 2 * Dm * 2);
  float* gates = (float*)alloc((size_t)B * 4 * Dm * 4);

  auto gemm = [&](const bf* A, int lda, const bf* W, int ldw,
                  float* Cf, bf* Cbf, int ldc, const float* bias,
                  int M, int Nn, int K, float alpha, int accum) {
    dim3 grid(Nn / 128, M / 16);
    k_gemm_bf16_wmma<<<grid, 256, 0, stream>>>(A, lda, W, ldw, Cf, Cbf, ldc,
                                               bias, M, Nn, K, alpha, accum);
  };

  // --- bf16 conversions (weights hit L2 once; negligible traffic) ---
  k_cvt_f32_bf16<<<2048, 256, 0, stream>>>(h_in, hbf, (size_t)B * Dm);
  k_cvt_f32_bf16<<<2048, 256, 0, stream>>>(in_proj_w, wqbf, (size_t)Dm * Dm);
  k_cvt_f32_bf16<<<2048, 256, 0, stream>>>(in_proj_w + (size_t)2 * Dm * Dm, wvbf,
                                           (size_t)Dm * Dm);
  k_transpose_wk<<<2048, 256, 0, stream>>>(in_proj_w, wkT);
  k_cvt_f32_bf16<<<2048, 256, 0, stream>>>(out_proj_w, wobf, (size_t)Dm * Dm);
  k_cvt_f32_bf16<<<4096, 256, 0, stream>>>(w_ih, wihbf, (size_t)4 * Dm * 2 * Dm);
  k_cvt_f32_bf16<<<4096, 256, 0, stream>>>(w_hh, whhbf, (size_t)4 * Dm * Dm);

  // --- q = h @ Wq^T + bq (bf16 out) ---
  gemm(hbf, Dm, wqbf, Dm, nullptr, qbf, Dm, in_proj_b, B, Dm, Dm, 1.0f, 0);

  // --- wkq[b,h,:] = (q_h @ Wk_h) / sqrt(DH) (bk drops out of softmax) ---
  const float inv_sqrt_dh = 0.044194173824159216f;  // 1/sqrt(512)
  for (int hh = 0; hh < 2; ++hh)
    gemm(qbf + hh * DHm, Dm, wkT + (size_t)hh * Dm * DHm, DHm,
         wkq + hh * Dm, nullptr, 2 * Dm, nullptr, B, Dm, DHm, inv_sqrt_dh, 0);

  // --- fused streaming attention: sv[b,h,:] = sum_n softmax * states ---
  k_attn_stream<<<B, 256, 0, stream>>>(all_states, wkq, svbf, N);

  // --- ctx[b,h,:] = sv_h @ Wv_h^T + bv_h ---
  for (int hh = 0; hh < 2; ++hh)
    gemm(svbf + hh * Dm, 2 * Dm, wvbf + (size_t)hh * DHm * Dm, Dm,
         ctxf + hh * DHm, ctxbf + hh * DHm, Dm,
         in_proj_b + 2 * Dm + hh * DHm, B, DHm, Dm, 1.0f, 0);

  // --- context = ctx @ out_proj_w^T + out_proj_b ---
  gemm(ctxbf, Dm, wobf, Dm, conf, nullptr, Dm, out_proj_b, B, Dm, Dm, 1.0f, 0);

  // --- gates = [context, ext] @ w_ih^T + b_ih + h @ w_hh^T + b_hh ---
  k_build_rnnin<<<2048, 256, 0, stream>>>(conf, ext_in, rnninbf, B);
  gemm(rnninbf, 2 * Dm, wihbf, 2 * Dm, gates, nullptr, 4 * Dm, b_ih,
       B, 4 * Dm, 2 * Dm, 1.0f, 0);
  gemm(hbf, Dm, whhbf, Dm, gates, nullptr, 4 * Dm, b_hh,
       B, 4 * Dm, Dm, 1.0f, 1);

  // --- LSTM cell + layernorm -> d_out ---
  k_lstm_ln<<<B, 256, 0, stream>>>(gates, c_in, ln_w, ln_b, out, B);
}
